// ClusteredAttention_23347442221162
// MI455X (gfx1250) — matrix operand order repaired and checked by reading, not compiled, not run on hardware
//
#include <hip/hip_runtime.h>
#include <hip/hip_bf16.h>

// ---------------------------------------------------------------------------
// Problem constants (from the reference)
// ---------------------------------------------------------------------------
#define NB   4
#define LSEQ 4096
#define DMOD 1024
#define NH   16
#define EH   64
#define NC   256
#define KITERS 10
#define NBITS 32
#define TEMP 0.125f

#define ML (NB * LSEQ)      // 16384 rows for the big GEMMs
#define BH (NB * NH)        // 64 batch*head slices

typedef __attribute__((ext_vector_type(16))) __bf16 v16bf;
typedef __attribute__((ext_vector_type(8)))  __bf16 v8bf;
typedef __attribute__((ext_vector_type(8)))  float  v8f;

// round-to-nearest-even f32 -> bf16 without relying on cast support
static __device__ __forceinline__ __bf16 f2bf(float f) {
    unsigned u = __builtin_bit_cast(unsigned, f);
    unsigned r = (u + 0x7FFFu + ((u >> 16) & 1u)) >> 16;
    unsigned short s = (unsigned short)r;
    return __builtin_bit_cast(__bf16, s);
}

// ---------------------------------------------------------------------------
// WMMA fragment loaders (CDNA5 wave32, V_WMMA_F32_16X16X32_BF16)
//
// A-matrix 16x32 (dense 16-bit table, ISA 7.12.2):
//   lane L: row M = L%16, h = L/16
//   VGPR0..3 hold K = 8h + {0..7}, VGPR4..7 hold K = 16 + 8h + {0..7}
//   -> two contiguous 8-element (16B) runs per lane.
// ---------------------------------------------------------------------------
static __device__ __forceinline__ v16bf load_fragA(const __bf16* base, int row0,
                                                   int k0, int ld) {
    int lane = threadIdx.x & 31;
    int m = lane & 15, h = lane >> 4;
    const __bf16* p = base + (size_t)(row0 + m) * ld + k0 + h * 8;
    v8bf lo = *reinterpret_cast<const v8bf*>(p);
    v8bf hi = *reinterpret_cast<const v8bf*>(p + 16);
    return __builtin_shufflevector(lo, hi, 0, 1, 2, 3, 4, 5, 6, 7,
                                   8, 9, 10, 11, 12, 13, 14, 15);
}

// B-matrix 32x16 (same K order as A):
//   lane L: col N = L%16, h = L/16; elements 0..7 hold
//   K = 8h + {0..7}, elements 8..15 hold K = 16 + 8h + {0..7}.
//   `base` is the transposed operand Bt[N][K] row-major.
static __device__ __forceinline__ v16bf load_fragB(const __bf16* base, int n0,
                                                   int k0, int ld) {
    int lane = threadIdx.x & 31;
    int nn = lane & 15, h = lane >> 4;
    const __bf16* p = base + (size_t)(n0 + nn) * ld + k0 + h * 8;
    v8bf lo = *reinterpret_cast<const v8bf*>(p);
    v8bf hi = *reinterpret_cast<const v8bf*>(p + 16);
    return __builtin_shufflevector(lo, hi, 0, 1, 2, 3, 4, 5, 6, 7,
                                   8, 9, 10, 11, 12, 13, 14, 15);
}

static __device__ __forceinline__ v8f wmma_bf16(v16bf a, v16bf b, v8f c) {
    return __builtin_amdgcn_wmma_f32_16x16x32_bf16(false, a, false, b,
                                                   (short)0, c, false, false);
}

// ---------------------------------------------------------------------------
// Conversion helpers
// ---------------------------------------------------------------------------
__global__ void cvt_f32_bf16_kernel(const float* __restrict__ src,
                                    __bf16* __restrict__ dst, size_t n) {
    size_t i = (size_t)blockIdx.x * blockDim.x + threadIdx.x;
    if (i < n) dst[i] = f2bf(src[i]);
}

// W [K=1024][N=1024] f32 row-major  ->  Wt [N][K] bf16
__global__ void transpose_wt_kernel(const float* __restrict__ W,
                                    __bf16* __restrict__ Wt) {
    size_t i = (size_t)blockIdx.x * blockDim.x + threadIdx.x; // over 1024*1024
    int kk = (int)(i & 1023);
    int n  = (int)(i >> 10);
    Wt[(size_t)n * 1024 + kk] = f2bf(W[(size_t)kk * 1024 + n]);
}

// ---------------------------------------------------------------------------
// GEMM: C[M][Nw] = A_bf16[M][K] * Bt_bf16[Nw][K]^T + bias, f32 out.
// One wave per 32x64 C tile (8 accumulators). Software pipelined: the next
// k-step's 6 fragment loads are issued BEFORE the current 8 WMMAs so the
// s_wait before the WMMAs is partial and VMEM latency overlaps XDL work.
// ---------------------------------------------------------------------------
__global__ __launch_bounds__(32)
void gemm_bf16_kernel(const __bf16* __restrict__ A, const __bf16* __restrict__ Bt,
                      const float* __restrict__ bias, float* __restrict__ C,
                      int M, int Nw, int K) {
    int n0 = blockIdx.x * 64;
    int m0 = blockIdx.y * 32;
    v8f acc[8] = {};

    v16bf a0 = load_fragA(A, m0, 0, K);
    v16bf a1 = load_fragA(A, m0 + 16, 0, K);
    v16bf b0 = load_fragB(Bt, n0, 0, K);
    v16bf b1 = load_fragB(Bt, n0 + 16, 0, K);
    v16bf b2 = load_fragB(Bt, n0 + 32, 0, K);
    v16bf b3 = load_fragB(Bt, n0 + 48, 0, K);

    for (int kk = 0; kk < K; kk += 32) {
        // issue next-step loads first (redundant re-load on the last step)
        int kn = (kk + 32 < K) ? kk + 32 : kk;
        v16bf na0 = load_fragA(A, m0, kn, K);
        v16bf na1 = load_fragA(A, m0 + 16, kn, K);
        v16bf nb0 = load_fragB(Bt, n0, kn, K);
        v16bf nb1 = load_fragB(Bt, n0 + 16, kn, K);
        v16bf nb2 = load_fragB(Bt, n0 + 32, kn, K);
        v16bf nb3 = load_fragB(Bt, n0 + 48, kn, K);

        acc[0] = wmma_bf16(a0, b0, acc[0]);
        acc[1] = wmma_bf16(a0, b1, acc[1]);
        acc[2] = wmma_bf16(a0, b2, acc[2]);
        acc[3] = wmma_bf16(a0, b3, acc[3]);
        acc[4] = wmma_bf16(a1, b0, acc[4]);
        acc[5] = wmma_bf16(a1, b1, acc[5]);
        acc[6] = wmma_bf16(a1, b2, acc[6]);
        acc[7] = wmma_bf16(a1, b3, acc[7]);

        a0 = na0; a1 = na1;
        b0 = nb0; b1 = nb1; b2 = nb2; b3 = nb3;
    }

    int lane = threadIdx.x & 31;
    int nn = lane & 15, hh = lane >> 4;
#pragma unroll
    for (int u = 0; u < 2; ++u) {
#pragma unroll
        for (int t = 0; t < 4; ++t) {
            int col = n0 + 16 * t + nn;
            float bv = bias[col];
#pragma unroll
            for (int r = 0; r < 8; ++r) {
                int row = m0 + 16 * u + r + 8 * hh;
                C[(size_t)row * Nw + col] = acc[4 * u + t][r] + bv;
            }
        }
    }
}

// ---------------------------------------------------------------------------
// LSH codes: one wave per (b,l); lane = hash bit; 32-bit code via OR-reduce.
// ---------------------------------------------------------------------------
__global__ __launch_bounds__(256)
void codes_kernel(const float* __restrict__ q, const float* __restrict__ planes,
                  unsigned* __restrict__ codes) {
    int wid = blockIdx.x * (blockDim.x >> 5) + (threadIdx.x >> 5); // b*LSEQ + l
    int b = wid >> 12, l = wid & (LSEQ - 1);
    int n = b >> 4, h = b & 15;
    const float* qr = q + ((size_t)(n * LSEQ + l)) * DMOD + h * EH;
    int bit = threadIdx.x & 31;
    float d = 0.f;
#pragma unroll 8
    for (int e = 0; e < EH; ++e) d += qr[e] * planes[e * NBITS + bit];
    unsigned m = (d > 0.f) ? (1u << bit) : 0u;
#pragma unroll
    for (int w = 1; w < 32; w <<= 1) m |= __shfl_xor(m, w, 32);
    if (bit == 0) codes[wid] = m;
}

// ---------------------------------------------------------------------------
// Hamming k-means: one block per (b); all state in LDS.
//   dist = csum + centsum - 2*dot  ==  popcount(code ^ cent) for binary data.
// ---------------------------------------------------------------------------
__global__ __launch_bounds__(256)
void kmeans_kernel(const unsigned* __restrict__ codes, int* __restrict__ assign) {
    __shared__ unsigned codes_s[LSEQ];
    __shared__ unsigned cent_s[NC];
    __shared__ unsigned short assign_s[LSEQ];
    int b = blockIdx.x, tid = threadIdx.x;
    const unsigned* cb = codes + (size_t)b * LSEQ;
    for (int l = tid; l < LSEQ; l += 256) codes_s[l] = cb[l];
    __syncthreads();
    // cent0 = codes[linspace(0, L-1, C).astype(int32)]
    cent_s[tid] = codes_s[(tid * (LSEQ - 1)) / (NC - 1)];
    __syncthreads();

    for (int it = 0; it < KITERS; ++it) {
        for (int l = tid; l < LSEQ; l += 256) {
            unsigned code = codes_s[l];
            int best = 1 << 30, bc = 0;
            for (int c = 0; c < NC; ++c) {
                int d = __popc(code ^ cent_s[c]);
                if (d < best) { best = d; bc = c; }   // '<' keeps first min (argmin ties)
            }
            assign_s[l] = (unsigned short)bc;
        }
        __syncthreads();
        // thread `tid` owns centroid `tid`: majority vote per bit
        int cnt = 0, bits[NBITS];
#pragma unroll
        for (int i = 0; i < NBITS; ++i) bits[i] = 0;
        for (int l = 0; l < LSEQ; ++l) {
            if (assign_s[l] == tid) {
                ++cnt;
                unsigned u = codes_s[l];
#pragma unroll
                for (int i = 0; i < NBITS; ++i) bits[i] += (u >> i) & 1u;
            }
        }
        if (cnt > 0) {
            unsigned nc = 0;
#pragma unroll
            for (int i = 0; i < NBITS; ++i)
                if (2 * bits[i] >= cnt) nc |= (1u << i);  // mean >= 0.5
            cent_s[tid] = nc;
        }
        __syncthreads();
    }
    // final assignment with final centroids
    int* ab = assign + (size_t)b * LSEQ;
    for (int l = tid; l < LSEQ; l += 256) {
        unsigned code = codes_s[l];
        int best = 1 << 30, bc = 0;
        for (int c = 0; c < NC; ++c) {
            int d = __popc(code ^ cent_s[c]);
            if (d < best) { best = d; bc = c; }
        }
        ab[l] = bc;
    }
}

// ---------------------------------------------------------------------------
// Cluster-mean queries: block per (c,b), thread per e. Deterministic (no atomics).
// ---------------------------------------------------------------------------
__global__ __launch_bounds__(64)
void qc_kernel(const float* __restrict__ q, const int* __restrict__ assign,
               float* __restrict__ qc, __bf16* __restrict__ qcbf) {
    int c = blockIdx.x, b = blockIdx.y, e = threadIdx.x;
    int n = b >> 4, h = b & 15;
    const int* ab = assign + (size_t)b * LSEQ;
    const float* qb = q + (size_t)n * LSEQ * DMOD + h * EH + e;
    float acc = 0.f;
    int cnt = 0;
    for (int l = 0; l < LSEQ; ++l) {
        if (ab[l] == c) { acc += qb[(size_t)l * DMOD]; ++cnt; }
    }
    float m = acc / (float)(cnt < 1 ? 1 : cnt);
    size_t o = ((size_t)b * NC + c) * EH + e;
    qc[o] = m;
    qcbf[o] = f2bf(m);
}

// ---------------------------------------------------------------------------
// Pack K (b,s,e) bf16 and V^T (b,e,s) bf16 from the [N*L, H*E] projections.
// ---------------------------------------------------------------------------
__global__ __launch_bounds__(256)
void pack_kv_kernel(const float* __restrict__ k, const float* __restrict__ v,
                    __bf16* __restrict__ kbf, __bf16* __restrict__ vtb) {
    size_t i = (size_t)blockIdx.x * blockDim.x + threadIdx.x; // BH*LSEQ*EH
    int e = (int)(i & (EH - 1));
    int s = (int)((i >> 6) & (LSEQ - 1));
    int b = (int)(i >> 18);
    int n = b >> 4, h = b & 15;
    size_t src = ((size_t)(n * LSEQ + s)) * DMOD + h * EH + e;
    kbf[i] = f2bf(k[src]);
    vtb[((size_t)b * EH + e) * LSEQ + s] = f2bf(v[src]);
}

// ---------------------------------------------------------------------------
// Clustered attention (flash style): one wave per (b, 16-cluster tile).
//   S = qc @ K^T * TEMP (WMMA), online softmax over 4096 keys,
//   O += P @ V (WMMA), P converted C-layout -> A-layout via 1KB LDS bounce.
//   Next K/V chunk prefetched (global_prefetch_b8) while current is computed.
// ---------------------------------------------------------------------------
__global__ __launch_bounds__(32)
void attn_kernel(const __bf16* __restrict__ qcbf, const __bf16* __restrict__ kbf,
                 const __bf16* __restrict__ vtb, float* __restrict__ vc) {
    int b = blockIdx.y;
    int c0 = blockIdx.x * 16;
    const __bf16* Q  = qcbf + (size_t)b * NC * EH;
    const __bf16* Kb = kbf  + (size_t)b * LSEQ * EH;
    const __bf16* Vb = vtb  + (size_t)b * EH * LSEQ;

    v16bf aQ0 = load_fragA(Q, c0, 0, EH);
    v16bf aQ1 = load_fragA(Q, c0, 32, EH);

    v8f o[4] = {};
    float mrow[8], lrow[8];
#pragma unroll
    for (int r = 0; r < 8; ++r) { mrow[r] = -INFINITY; lrow[r] = 0.f; }

    __shared__ __align__(16) __bf16 Pl[16 * 32];
    int lane = threadIdx.x & 31;
    int nn = lane & 15, hh = lane >> 4;

    for (int s0 = 0; s0 < LSEQ; s0 += 32) {
        // prefetch next chunk: 32 K-rows (one per lane) + 64 V^T row segments
        int sp = s0 + 32;
        if (sp < LSEQ) {
            __builtin_prefetch(Kb + (size_t)(sp + lane) * EH, 0, 3);
            __builtin_prefetch(Vb + (size_t)lane * LSEQ + sp, 0, 3);
            __builtin_prefetch(Vb + (size_t)(lane + 32) * LSEQ + sp, 0, 3);
        }

        v8f sacc[2] = {};
#pragma unroll
        for (int half = 0; half < 2; ++half) {
            v16bf bk0 = load_fragB(Kb, s0 + 16 * half, 0, EH);
            sacc[half] = wmma_bf16(aQ0, bk0, sacc[half]);
            v16bf bk1 = load_fragB(Kb, s0 + 16 * half, 32, EH);
            sacc[half] = wmma_bf16(aQ1, bk1, sacc[half]);
        }
        float p0[8], p1[8];
#pragma unroll
        for (int r = 0; r < 8; ++r) {
            float x0 = sacc[0][r] * TEMP;
            float x1 = sacc[1][r] * TEMP;
            float mx = fmaxf(x0, x1);
#pragma unroll
            for (int w = 1; w < 16; w <<= 1) mx = fmaxf(mx, __shfl_xor(mx, w, 32));
            float mnew = fmaxf(mrow[r], mx);
            float scale = expf(mrow[r] - mnew);
            float e0 = expf(x0 - mnew);
            float e1 = expf(x1 - mnew);
            float ps = e0 + e1;
#pragma unroll
            for (int w = 1; w < 16; w <<= 1) ps += __shfl_xor(ps, w, 32);
            lrow[r] = lrow[r] * scale + ps;
            mrow[r] = mnew;
#pragma unroll
            for (int t = 0; t < 4; ++t) o[t][r] *= scale;
            p0[r] = e0;
            p1[r] = e1;
        }
        __syncthreads();
#pragma unroll
        for (int r = 0; r < 8; ++r) {
            Pl[(r + 8 * hh) * 32 + nn]      = f2bf(p0[r]);
            Pl[(r + 8 * hh) * 32 + 16 + nn] = f2bf(p1[r]);
        }
        __syncthreads();
        v16bf aP = load_fragA((const __bf16*)Pl, 0, 0, 32);
#pragma unroll
        for (int t = 0; t < 4; ++t) {
            v16bf bV = load_fragB(Vb, 16 * t, s0, LSEQ);
            o[t] = wmma_bf16(aP, bV, o[t]);
        }
    }
#pragma unroll
    for (int t = 0; t < 4; ++t) {
#pragma unroll
        for (int r = 0; r < 8; ++r) {
            int c = c0 + r + 8 * hh;
            vc[((size_t)b * NC + c) * EH + 16 * t + nn] = o[t][r] / lrow[r];
        }
    }
}

// ---------------------------------------------------------------------------
// Scatter cluster outputs back to tokens -> bf16 [N*L, H*E] for final GEMM.
// ---------------------------------------------------------------------------
__global__ __launch_bounds__(256)
void gather_kernel(const float* __restrict__ vc, const int* __restrict__ assign,
                   __bf16* __restrict__ attnbf) {
    size_t i = (size_t)blockIdx.x * blockDim.x + threadIdx.x; // ML*DMOD
    int e = (int)(i & (EH - 1));
    int h = (int)((i >> 6) & (NH - 1));
    size_t nl = i >> 10;
    int l = (int)(nl & (LSEQ - 1));
    int n = (int)(nl >> 12);
    int b = n * NH + h;
    int a = assign[(size_t)b * LSEQ + l];
    attnbf[i] = f2bf(vc[((size_t)b * NC + a) * EH + e]);
}

// ---------------------------------------------------------------------------
// Host orchestration
// ---------------------------------------------------------------------------
extern "C" void kernel_launch(void* const* d_in, const int* in_sizes, int n_in,
                              void* d_out, int out_size, void* d_ws, size_t ws_size,
                              hipStream_t stream) {
    (void)in_sizes; (void)n_in; (void)out_size; (void)ws_size;
    const float* x      = (const float*)d_in[0];
    const float* Wq     = (const float*)d_in[2];
    const float* bq     = (const float*)d_in[3];
    const float* Wk     = (const float*)d_in[4];
    const float* bk     = (const float*)d_in[5];
    const float* Wv     = (const float*)d_in[6];
    const float* bv     = (const float*)d_in[7];
    const float* Wo     = (const float*)d_in[8];
    const float* bo     = (const float*)d_in[9];
    const float* planes = (const float*)d_in[10];

    char* ws = (char*)d_ws;
    size_t off = 0;
    auto alloc = [&](size_t bytes) -> void* {
        void* p = ws + off;
        off += (bytes + 255) & ~(size_t)255;
        return p;
    };

    float*    qf     = (float*)alloc((size_t)ML * DMOD * 4);
    float*    kf     = (float*)alloc((size_t)ML * DMOD * 4);
    float*    vf     = (float*)alloc((size_t)ML * DMOD * 4);
    __bf16*   xbf    = (__bf16*)alloc((size_t)ML * DMOD * 2);
    __bf16*   wqt    = (__bf16*)alloc((size_t)DMOD * DMOD * 2);
    __bf16*   wkt    = (__bf16*)alloc((size_t)DMOD * DMOD * 2);
    __bf16*   wvt    = (__bf16*)alloc((size_t)DMOD * DMOD * 2);
    __bf16*   wot    = (__bf16*)alloc((size_t)DMOD * DMOD * 2);
    unsigned* codes  = (unsigned*)alloc((size_t)BH * LSEQ * 4);
    int*      assign = (int*)alloc((size_t)BH * LSEQ * 4);
    float*    qcf    = (float*)alloc((size_t)BH * NC * EH * 4);
    __bf16*   qcbf   = (__bf16*)alloc((size_t)BH * NC * EH * 2);
    __bf16*   kbf    = (__bf16*)alloc((size_t)BH * LSEQ * EH * 2);
    __bf16*   vtb    = (__bf16*)alloc((size_t)BH * EH * LSEQ * 2);
    float*    vcf    = (float*)alloc((size_t)BH * NC * EH * 4);
    __bf16*   attnbf = (__bf16*)alloc((size_t)ML * DMOD * 2);

    // 1) bf16 conversions of activations and (transposed) weights
    cvt_f32_bf16_kernel<<<(ML * DMOD) / 256, 256, 0, stream>>>(x, xbf, (size_t)ML * DMOD);
    transpose_wt_kernel<<<(DMOD * DMOD) / 256, 256, 0, stream>>>(Wq, wqt);
    transpose_wt_kernel<<<(DMOD * DMOD) / 256, 256, 0, stream>>>(Wk, wkt);
    transpose_wt_kernel<<<(DMOD * DMOD) / 256, 256, 0, stream>>>(Wv, wvt);
    transpose_wt_kernel<<<(DMOD * DMOD) / 256, 256, 0, stream>>>(Wo, wot);

    // 2) QKV projections (WMMA, 32x64 tile per wave, pipelined)
    dim3 ggrid(DMOD / 64, ML / 32);
    gemm_bf16_kernel<<<ggrid, 32, 0, stream>>>(xbf, wqt, bq, qf, ML, DMOD, DMOD);
    gemm_bf16_kernel<<<ggrid, 32, 0, stream>>>(xbf, wkt, bk, kf, ML, DMOD, DMOD);
    gemm_bf16_kernel<<<ggrid, 32, 0, stream>>>(xbf, wvt, bv, vf, ML, DMOD, DMOD);

    // 3) LSH codes (one wave per token-head)
    codes_kernel<<<(BH * LSEQ) / 8, 256, 0, stream>>>(qf, planes, codes);

    // 4) Hamming k-means, all in LDS
    kmeans_kernel<<<BH, 256, 0, stream>>>(codes, assign);

    // 5) cluster-mean queries
    qc_kernel<<<dim3(NC, BH), 64, 0, stream>>>(qf, assign, qcf, qcbf);

    // 6) pack K / V^T to bf16 WMMA-friendly layouts
    pack_kv_kernel<<<((size_t)BH * LSEQ * EH) / 256, 256, 0, stream>>>(kf, vf, kbf, vtb);

    // 7) clustered attention (WMMA + online softmax + prefetch)
    attn_kernel<<<dim3(NC / 16, BH), 32, 0, stream>>>(qcbf, kbf, vtb, vcf);

    // 8) scatter back to tokens
    gather_kernel<<<(ML * DMOD) / 256, 256, 0, stream>>>(vcf, assign, attnbf);

    // 9) output projection (WMMA) -> d_out f32
    gemm_bf16_kernel<<<ggrid, 32, 0, stream>>>(attnbf, wot, bo, (float*)d_out, ML, DMOD, DMOD);
}
